// Decoder_32744830664720
// MI455X (gfx1250) — matrix operand order repaired and checked by reading, not compiled, not run on hardware
//
#include <hip/hip_runtime.h>
#include <math.h>

// ---------------------------------------------------------------------------
// MI455X (gfx1250) Tacotron decoder step.
// Dominant op: enc_out[64,1024,512] @ att_Wv[512,1024]  (68.7 GFLOP) fused with
// tanh(q+v)@att_v -> score.  HBM-bound (134MB enc read @23.3TB/s ~ 5.8us):
//  - enc tile staged ONCE to LDS as bf16, reused over all N (no 268MB v tensor)
//  - att_Wv pre-converted to bf16 AND pre-transposed to [n][k] so the Tensor
//    Data Mover can restage each 64-col chunk verbatim (tensor_load_to_lds),
//    double-buffered and overlapped with WMMA via s_wait_tensorcnt.
//  - TDM pad feature (pad_interval=7 -> every 256 DWORDs, pad_amount=4 -> 5
//    DWORDs) reproduces the bank-conflict-free LDS pitch of 522 ushorts.
//  - matrix math via v_wmma_f32_16x16x32_bf16 (wave32 CDNA5 WMMA)
// LSTM trick: reference uses h0=c0=0 -> zh=0, so gates = beta1*(x@W)+b and both
// [1024x4096] U-matmuls vanish exactly.
// ---------------------------------------------------------------------------

typedef __attribute__((ext_vector_type(16))) __bf16 v16bf;
typedef __attribute__((ext_vector_type(8)))  float  v8f;

#define NB      64      // batch
#define TENC    1024
#define ENCD    512
#define UNITS   1024
#define PRE     256
#define FRAME   160
#define G4      4096    // 4*UNITS
#define LIN     768     // ENCD + PRE
#define PIN     1536    // ENCD + UNITS
#define ZO      0.1f

// bank-conflict-free LDS pitches (pitch*2B/4B % 64 coprime with 64)
#define PK3 522         // attn kernel: 512 + 10  (== TDM pad 5 DWORDs/row)
#define PG  138         // gemm kernel: 128 + 10

// TDM path only in the device pass and only if the toolchain declares it.
#if defined(__HIP_DEVICE_COMPILE__) &&                                         \
    __has_builtin(__builtin_amdgcn_tensor_load_to_lds) &&                      \
    __has_builtin(__builtin_amdgcn_s_wait_tensorcnt)
#define HAVE_TDM 1
#else
#define HAVE_TDM 0
#endif

__device__ __forceinline__ unsigned short f2bf(float f) {
  union { float f; unsigned u; } x; x.f = f;
  unsigned r = x.u + 0x7FFFu + ((x.u >> 16) & 1u);   // round-to-nearest-even
  return (unsigned short)(r >> 16);
}
__device__ __forceinline__ float sigf(float x) { return 1.0f / (1.0f + __expf(-x)); }

union AFrag { v16bf v; unsigned u[8]; };

__device__ __forceinline__ v8f wmma_bf16(v16bf a, v16bf b, v8f c) {
  // (neg_a, A, neg_b, B, c_mod, C, reuse_a, reuse_b)
  return __builtin_amdgcn_wmma_f32_16x16x32_bf16(false, a, false, b, (short)0, c,
                                                 false, false);
}

#if HAVE_TDM
// Issue one TDM transfer: 2D tile [64 rows x 512 bf16] from wvT (row-major,
// row stride 512) into LDS at ldsOff with 20B padding after each 1024B row
// (net LDS pitch = 522 ushorts = PK3).  D# layout per CDNA5 ISA §8.3/8.4.
// This toolchain's builtin is the 6-arg clang-23 form:
//   (uint32x4 g0, int32x8 g1, int32x4 g2, int32x4 g3, int32x8 g4, i32 cpol)
__device__ __forceinline__ void tdm_load_chunk(const unsigned short* gsrc,
                                               unsigned ldsOff) {
  typedef __attribute__((ext_vector_type(4))) unsigned int u32x4;
  typedef __attribute__((ext_vector_type(8))) int i32x8;
  typedef __attribute__((ext_vector_type(4))) int i32x4;
  unsigned long long ga = (unsigned long long)(size_t)gsrc;
  u32x4 g0;
  g0[0] = 1u;                                          // count=1 (user mode)
  g0[1] = ldsOff;                                      // lds_addr (bytes)
  g0[2] = (unsigned)(ga & 0xFFFFFFFFu);                // global_addr[31:0]
  g0[3] = (unsigned)((ga >> 32) & 0x01FFFFFFu) | (2u << 30);  // [56:32]|type=2
  i32x8 g1;
  g1[0] = (1 << 16)      // data_size = 2 bytes
        | (1 << 20)      // pad_enable
        | (7 << 22)      // pad_interval: every 256 DWORDs (= one 1024B row)
        | (4 << 25);     // pad_amount: 5 DWORDs (= 10 ushorts -> pitch 522)
  g1[1] = (int)(512u << 16);   // tensor_dim0 = 512 (k extent)
  g1[2] = (int)(64u << 16);    // tensor_dim0 hi=0 | tensor_dim1 = 64 (rows)
  g1[3] = (int)(512u << 16);   // tensor_dim1 hi=0 | tile_dim0 = 512
  g1[4] = 64;                  // tile_dim1 = 64, tile_dim2 = 0
  g1[5] = 512;                 // tensor_dim0_stride[31:0] = 512 elements
  g1[6] = 0;                   // stride0 hi | tensor_dim1_stride lo
  g1[7] = 0;
  i32x4 z4 = {0, 0, 0, 0};
  i32x8 z8 = {0, 0, 0, 0, 0, 0, 0, 0};
  __builtin_amdgcn_tensor_load_to_lds(g0, g1, z4, z4, z8, 0);
}
#endif

// --- K1: prenet + query (biases bq+bv folded into qb) -----------------------
__global__ void prenet_kernel(const float* __restrict__ prev,
                              const float* __restrict__ p1W, const float* __restrict__ p1b,
                              const float* __restrict__ p2W, const float* __restrict__ p2b,
                              const float* __restrict__ Wq,  const float* __restrict__ bq,
                              const float* __restrict__ bv,
                              float* __restrict__ xout, float* __restrict__ qb) {
  __shared__ float pv[FRAME];
  __shared__ float x1[PRE];
  __shared__ float x2[PRE];
  const int b = blockIdx.x, tid = threadIdx.x;
  if (tid < FRAME) pv[tid] = prev[b * FRAME + tid];
  __syncthreads();
  {
    float a = p1b[tid];
    for (int k = 0; k < FRAME; ++k) a += pv[k] * p1W[k * PRE + tid];
    x1[tid] = fmaxf(a, 0.0f);
  }
  __syncthreads();
  {
    float a = p2b[tid];
    for (int k = 0; k < PRE; ++k) a += x1[k] * p2W[k * PRE + tid];
    float r = fmaxf(a, 0.0f);
    x2[tid] = r;
    xout[b * PRE + tid] = r;
  }
  __syncthreads();
  for (int n = tid; n < UNITS; n += 256) {
    float a = bq[n] + bv[n];
    for (int k = 0; k < PRE; ++k) a += x2[k] * Wq[k * UNITS + n];
    qb[b * UNITS + n] = a;
  }
}

// --- K2: att_Wv fp32 [k][n] -> bf16 transposed [n][k] (TDM/L2-friendly) -----
__global__ void cvt_wv_kernel(const float* __restrict__ src,
                              unsigned short* __restrict__ wvT) {
  int i = blockIdx.x * 256 + threadIdx.x;            // i = n*512 + k
  if (i < ENCD * UNITS) {
    int n = i >> 9, k = i & (ENCD - 1);
    wvT[i] = f2bf(src[k * UNITS + n]);
  }
}

// --- K3: fused attention energies ------------------------------------------
// Grid (TENC/64, NB). Block 256 (8 waves). Per WG: stage enc rows [64 x 512]
// once (bf16, LDS), then sweep N=1024 in 64-col chunks of WvT; chunks are
// double-buffered via the Tensor Data Mover so DMA overlaps WMMA.
// score[t] = sum_n att_v[n]*tanh(qb[n] + v[t,n]) accumulated in registers.
__global__ void attn_score_kernel(const float* __restrict__ enc,
                                  const unsigned short* __restrict__ wvT,
                                  const float* __restrict__ qb,
                                  const float* __restrict__ attv,
                                  float* __restrict__ score) {
  extern __shared__ char smem[];
  unsigned short* sA  = (unsigned short*)smem;            // [64][PK3]
  unsigned short* sB0 = sA + 64 * PK3;                    // [64][PK3] (n-major)
  unsigned short* sB1 = sB0 + 64 * PK3;                   // double buffer
  float* sScore = (float*)(sB1 + 64 * PK3);               // [64]

  const int tid = threadIdx.x;
  const int w = tid >> 5, lane = tid & 31;
  const int b = blockIdx.y, tBase = blockIdx.x * 64;
  const int rowBase = 16 * (w & 3);        // waves 0-3 / 4-7 each cover 4 row tiles
  const int nsub    = 32 * (w >> 2);       // and one of two 32-col halves

  if (tid < 64) sScore[tid] = 0.0f;
  // stage enc rows once: [64 x 512] fp32 -> bf16 (the single HBM read of enc)
  for (int i = tid; i < 64 * ENCD; i += 256) {
    int r = i >> 9, k = i & (ENCD - 1);
    sA[r * PK3 + k] = f2bf(enc[((b << 10) + tBase + r) * ENCD + k]);
  }

#if HAVE_TDM
  if (tid == 0)                                      // prefetch chunk 0
    tdm_load_chunk(wvT, (unsigned)(size_t)sB0);
#endif

  float acc[8];
#pragma unroll
  for (int g = 0; g < 8; ++g) acc[g] = 0.0f;

  const int arow  = (rowBase + (lane & 15)) * PK3;
  const int kgrpA = (lane >> 4) << 3;                    // 0 or 8
  const int kgrpB = (lane >> 4) << 3;                    // 0 or 8

  for (int nc = 0; nc < UNITS / 64; ++nc) {              // 16 chunks of 64 cols
    unsigned short* sB = (nc & 1) ? sB1 : sB0;
#if HAVE_TDM
    if (tid == 0) {
      if (nc + 1 < UNITS / 64) {                         // prefetch next chunk
        tdm_load_chunk(wvT + (size_t)(nc + 1) * 64 * ENCD,
                       (unsigned)(size_t)((nc & 1) ? sB0 : sB1));
        __builtin_amdgcn_s_wait_tensorcnt((short)1);     // chunk nc landed
      } else {
        __builtin_amdgcn_s_wait_tensorcnt((short)0);
      }
    }
    __syncthreads();                                     // publish sB to all waves
#else
    __syncthreads();                                     // prev chunk consumed
    for (int i = tid; i < 64 * ENCD; i += 256) {         // sync fallback staging
      int n = i >> 9, k = i & (ENCD - 1);
      sB[n * PK3 + k] = wvT[(size_t)(nc * 64 + n) * ENCD + k];
    }
    __syncthreads();
#endif

    v8f c0 = {}, c1 = {};
    const int b0 = (nsub + (lane & 15)) * PK3;
    const int b1 = (nsub + 16 + (lane & 15)) * PK3;
#pragma unroll 4
    for (int kb = 0; kb < ENCD; kb += 32) {
      AFrag a, f0, f1;
#pragma unroll
      for (int i = 0; i < 4; ++i) {
        a.u[i]     = *(const unsigned*)&sA[arow + kb + kgrpA + 2 * i];
        a.u[4 + i] = *(const unsigned*)&sA[arow + kb + 16 + kgrpA + 2 * i];
      }
#pragma unroll
      for (int i = 0; i < 8; ++i) {
        f0.u[i] = *(const unsigned*)&sB[b0 + kb + kgrpB + 2 * i + ((i >> 2) << 3)];
        f1.u[i] = *(const unsigned*)&sB[b1 + kb + kgrpB + 2 * i + ((i >> 2) << 3)];
      }
      c0 = wmma_bf16(a.v, f0.v, c0);
      c1 = wmma_bf16(a.v, f1.v, c1);
    }
    // fused epilogue: score += att_v[n] * tanh(qb[n] + v[t,n])
    const int ng0 = nc * 64 + nsub + (lane & 15);
    const int ng1 = ng0 + 16;
    const float q0 = qb[b * UNITS + ng0], q1 = qb[b * UNITS + ng1];
    const float a0 = attv[ng0], a1 = attv[ng1];
#pragma unroll
    for (int g = 0; g < 8; ++g)
      acc[g] += tanhf(q0 + c0[g]) * a0 + tanhf(q1 + c1[g]) * a1;
    __syncthreads();                 // all reads of sB done before its refill
  }
  // cross-lane / cross-wave reduction of per-row partials
  const int rofs = rowBase + 8 * (lane >> 4);
#pragma unroll
  for (int g = 0; g < 8; ++g) atomicAdd(&sScore[rofs + g], acc[g]);
  __syncthreads();
  if (tid < 64) score[b * TENC + tBase + tid] = sScore[tid];
}

// --- K4: softmax over T + context, feed lstm_in / proj_in -------------------
__global__ void softmax_ctx_kernel(const float* __restrict__ score,
                                   const float* __restrict__ enc,
                                   const float* __restrict__ x,
                                   float* __restrict__ lstm_in,
                                   float* __restrict__ proj_in) {
  __shared__ float p[TENC];
  __shared__ float red[256];
  const int b = blockIdx.x, tid = threadIdx.x;
  float m = -3.4e38f;
  for (int i = tid; i < TENC; i += 256) { float v = score[b * TENC + i]; p[i] = v; m = fmaxf(m, v); }
  red[tid] = m; __syncthreads();
  for (int s = 128; s > 0; s >>= 1) { if (tid < s) red[tid] = fmaxf(red[tid], red[tid + s]); __syncthreads(); }
  m = red[0]; __syncthreads();
  float sum = 0.0f;
  for (int i = tid; i < TENC; i += 256) { float e = __expf(p[i] - m); p[i] = e; sum += e; }
  red[tid] = sum; __syncthreads();
  for (int s = 128; s > 0; s >>= 1) { if (tid < s) red[tid] += red[tid + s]; __syncthreads(); }
  const float inv = 1.0f / red[0];
  __syncthreads();
  // context: second enc pass (L2-resident: 134MB < 192MB L2)
  float a0 = 0.0f, a1 = 0.0f;
  for (int t = 0; t < TENC; ++t) {
    const float pt = p[t];
    const float* er = enc + ((size_t)(b << 10) + t) * ENCD;
    a0 += pt * er[tid];
    a1 += pt * er[tid + 256];
  }
  a0 *= inv; a1 *= inv;
  lstm_in[b * LIN + tid]        = a0;
  lstm_in[b * LIN + tid + 256]  = a1;
  proj_in[b * PIN + tid]        = a0;
  proj_in[b * PIN + tid + 256]  = a1;
  lstm_in[b * LIN + ENCD + tid] = x[b * PRE + tid];   // concat([context, x])
}

// --- K5/K7: generic M=64 bf16-WMMA GEMM: out[64,N] = A[64,K] @ Bm[K,N] ------
__global__ void gemm64_bf16_kernel(const float* __restrict__ A, int K,
                                   const float* __restrict__ Bm, int N,
                                   float* __restrict__ out) {
  extern __shared__ char smem[];
  unsigned short* sA = (unsigned short*)smem;   // [64][PG]
  unsigned short* sB = sA + 64 * PG;            // [64][PG] (n-major)
  const int tid = threadIdx.x;
  const int w = tid >> 5, lane = tid & 31;
  const int nBase = blockIdx.x * 64;
  const int rowBase = 16 * (w & 3);
  const int nsub    = 32 * (w >> 2);

  const int arow  = (rowBase + (lane & 15)) * PG;
  const int kgrpA = (lane >> 4) << 3;
  const int kgrpB = (lane >> 4) << 3;
  const int b0 = (nsub + (lane & 15)) * PG;
  const int b1 = (nsub + 16 + (lane & 15)) * PG;

  v8f c0 = {}, c1 = {};
  for (int kb0 = 0; kb0 < K; kb0 += 128) {
    __syncthreads();
    for (int i = tid; i < 64 * 128; i += 256) {          // stage A (fp32->bf16)
      int r = i >> 7, k = i & 127;
      sA[r * PG + k] = f2bf(A[r * K + kb0 + k]);
    }
    for (int i = tid; i < 64 * 128; i += 256) {          // stage B transposed
      int n = i & 63, k = i >> 6;
      sB[n * PG + k] = f2bf(Bm[(size_t)(kb0 + k) * N + nBase + n]);
    }
    if (kb0 + 128 < K)                                   // global_prefetch_b8
      __builtin_prefetch(&Bm[(size_t)(kb0 + 128 + (tid >> 6)) * N + nBase + (tid & 63)], 0, 1);
    __syncthreads();
#pragma unroll
    for (int kb = 0; kb < 128; kb += 32) {
      AFrag a, f0, f1;
#pragma unroll
      for (int i = 0; i < 4; ++i) {
        a.u[i]     = *(const unsigned*)&sA[arow + kb + kgrpA + 2 * i];
        a.u[4 + i] = *(const unsigned*)&sA[arow + kb + 16 + kgrpA + 2 * i];
      }
#pragma unroll
      for (int i = 0; i < 8; ++i) {
        f0.u[i] = *(const unsigned*)&sB[b0 + kb + kgrpB + 2 * i + ((i >> 2) << 3)];
        f1.u[i] = *(const unsigned*)&sB[b1 + kb + kgrpB + 2 * i + ((i >> 2) << 3)];
      }
      c0 = wmma_bf16(a.v, f0.v, c0);
      c1 = wmma_bf16(a.v, f1.v, c1);
    }
  }
  const int r0 = rowBase + 8 * (lane >> 4);
  const int n0 = nBase + nsub + (lane & 15), n1 = n0 + 16;
#pragma unroll
  for (int g = 0; g < 8; ++g) {
    out[(size_t)(r0 + g) * N + n0] = c0[g];
    out[(size_t)(r0 + g) * N + n1] = c1[g];
  }
}

// --- K6/K8: MI-LSTM gates with h0=c0=0 (zh=0 -> gates = beta1*zx + b) -------
__global__ void milstm_gate_kernel(const float* __restrict__ zx,
                                   const float* __restrict__ beta1,
                                   const float* __restrict__ bias,
                                   float* __restrict__ outH, int pitch, int off) {
  const int idx = blockIdx.x * 256 + threadIdx.x;
  const int b = idx >> 10, u = idx & (UNITS - 1);
  const float* zr = zx + (size_t)b * G4;
  const float gi = beta1[u]            * zr[u]            + bias[u];
  const float gc = beta1[2 * UNITS + u] * zr[2 * UNITS + u] + bias[2 * UNITS + u];
  const float go = beta1[3 * UNITS + u] * zr[3 * UNITS + u] + bias[3 * UNITS + u];
  const float cn = sigf(gi) * tanhf(gc);             // c0=0: f-gate term vanishes
  outH[(size_t)b * pitch + off + u] = (1.0f - ZO) * sigf(go) * tanhf(cn);
}

// --- K9: output projections -------------------------------------------------
__global__ void proj_kernel(const float* __restrict__ proj_in,
                            const float* __restrict__ fW, const float* __restrict__ fb,
                            const float* __restrict__ sW, const float* __restrict__ sb,
                            float* __restrict__ out) {
  __shared__ float row[PIN];
  const int b = blockIdx.x, tid = threadIdx.x;
  for (int i = tid; i < PIN; i += 256) row[i] = proj_in[b * PIN + i];
  __syncthreads();
  if (tid < FRAME) {
    float a = fb[tid];
    for (int k = 0; k < PIN; ++k) a += row[k] * fW[k * FRAME + tid];
    out[b * FRAME + tid] = a;
  } else if (tid == FRAME) {
    float a = sb[0];
    for (int k = 0; k < PIN; ++k) a += row[k] * sW[k];
    out[NB * FRAME + b] = sigf(a);                  // stop after frames block
  }
}

// ---------------------------------------------------------------------------
extern "C" void kernel_launch(void* const* d_in, const int* in_sizes, int n_in,
                              void* d_out, int out_size, void* d_ws, size_t ws_size,
                              hipStream_t stream) {
  const float* prev   = (const float*)d_in[0];
  const float* enc    = (const float*)d_in[1];
  const float* p1W    = (const float*)d_in[2];
  const float* p1b    = (const float*)d_in[3];
  const float* p2W    = (const float*)d_in[4];
  const float* p2b    = (const float*)d_in[5];
  const float* attWq  = (const float*)d_in[6];
  const float* attbq  = (const float*)d_in[7];
  const float* attWv  = (const float*)d_in[8];
  const float* attbv  = (const float*)d_in[9];
  const float* attv   = (const float*)d_in[10];
  const float* l1W    = (const float*)d_in[11];
  const float* l1b    = (const float*)d_in[13];
  const float* l1beta = (const float*)d_in[15];
  const float* l2W    = (const float*)d_in[17];
  const float* l2b    = (const float*)d_in[19];
  const float* l2beta = (const float*)d_in[21];
  const float* stopW  = (const float*)d_in[23];
  const float* stopb  = (const float*)d_in[24];
  const float* frW    = (const float*)d_in[25];
  const float* frb    = (const float*)d_in[26];
  float* out = (float*)d_out;

  // workspace layout (fp32 unless noted)
  float* wsf       = (float*)d_ws;
  float* x_pre     = wsf;                   // [64,256]
  float* qb        = x_pre   + NB * PRE;    // [64,1024]  q + bq + bv
  float* score     = qb      + NB * UNITS;  // [64,1024]
  float* lstm_in   = score   + NB * TENC;   // [64,768]
  float* zx        = lstm_in + NB * LIN;    // [64,4096]  (reused lstm1/lstm2)
  float* h1        = zx      + NB * G4;     // [64,1024]
  float* proj_in   = h1      + NB * UNITS;  // [64,1536]
  unsigned short* wvT = (unsigned short*)(proj_in + NB * PIN);  // [1024,512] bf16 (n-major)

  // shared sizes are path-independent (host pass can't see device builtins)
  const size_t shAttn = (size_t)3 * 64 * PK3 * 2 + 64 * 4;   // ~196 KB (320KB/WGP)
  const size_t shGemm = (size_t)2 * 64 * PG * 2;             // ~34.5 KB

  prenet_kernel<<<NB, 256, 0, stream>>>(prev, p1W, p1b, p2W, p2b, attWq, attbq,
                                        attbv, x_pre, qb);
  cvt_wv_kernel<<<(ENCD * UNITS + 255) / 256, 256, 0, stream>>>(attWv, wvT);
  attn_score_kernel<<<dim3(TENC / 64, NB), 256, shAttn, stream>>>(enc, wvT, qb,
                                                                  attv, score);
  softmax_ctx_kernel<<<NB, 256, 0, stream>>>(score, enc, x_pre, lstm_in, proj_in);
  // LSTM1: zx = [context|x][64,768] @ l1_W[768,4096]
  gemm64_bf16_kernel<<<G4 / 64, 256, shGemm, stream>>>(lstm_in, LIN, l1W, G4, zx);
  milstm_gate_kernel<<<NB * UNITS / 256, 256, 0, stream>>>(zx, l1beta, l1b, h1,
                                                           UNITS, 0);
  // LSTM2: zx = h1[64,1024] @ l2_W[1024,4096]; h2 written into proj_in[:,512:]
  gemm64_bf16_kernel<<<G4 / 64, 256, shGemm, stream>>>(h1, UNITS, l2W, G4, zx);
  milstm_gate_kernel<<<NB * UNITS / 256, 256, 0, stream>>>(zx, l2beta, l2b,
                                                           proj_in, PIN, ENCD);
  proj_kernel<<<NB, 256, 0, stream>>>(proj_in, frW, frb, stopW, stopb, out);
}
